// TextDeformableTransformerDecoderLayer_14276471292333
// MI455X (gfx1250) — matrix operand order repaired and checked by reading, not compiled, not run on hardware
//
#include <hip/hip_runtime.h>
#include <math.h>

// ---------------- problem constants (match reference) ----------------
#define BB   2
#define QQ   300
#define TT   25
#define CC   256
#define NHH  8
#define DFFF 1024
#define HDD  32
#define LQQ  (QQ * TT)        // 7500
#define SS   21760            // 128^2 + 64^2 + 32^2 + 16^2
#define M_TOK (BB * QQ * TT)  // 15000

typedef __attribute__((ext_vector_type(16))) __bf16 v16bf;
typedef __attribute__((ext_vector_type(2)))  __bf16 v2bf;
typedef __attribute__((ext_vector_type(8)))  float  v8f;

static __device__ __forceinline__ unsigned f2bf1(float x) {
  unsigned u = __float_as_uint(x);
  u += 0x7fffu + ((u >> 16) & 1u);   // round-to-nearest-even
  return u >> 16;
}
static __device__ __forceinline__ unsigned pk2(float lo, float hi) {
#if __has_builtin(__builtin_amdgcn_cvt_pk_bf16_f32)
  union { v2bf v; unsigned u; } cv;
  cv.v = __builtin_amdgcn_cvt_pk_bf16_f32(lo, hi);
  return cv.u;
#else
  return f2bf1(lo) | (f2bf1(hi) << 16);
#endif
}

// =====================================================================
// Generic GEMM: Y[M,N] = X[M,K] @ W[N,K]^T + bias, optional ReLU.
// bf16 inputs (converted on the fly), fp32 accumulate via
// v_wmma_f32_16x16x32_bf16. Block = 256 threads = 8 waves.
// Block tile 64x128, wave tile 16x64 (4 accumulators -> 4 WMMA/K-step).
// Requires: K % 32 == 0, N % 128 == 0. M edge-guarded.
// =====================================================================
__global__ __launch_bounds__(256) void gemm_bf16_kernel(
    const float* __restrict__ X, const float* __restrict__ W,
    const float* __restrict__ bias, float* __restrict__ Y,
    int M, int N, int K, int relu)
{
  // rows padded to 40 bf16 (80B): 16B-aligned b128 rows, conflict-free banks
  __shared__ __align__(16) unsigned short Xs[64 * 40];
  __shared__ __align__(16) unsigned short Ws2[128 * 40];

  const int tid   = threadIdx.x;
  const int lane  = tid & 31;
  const int wave  = tid >> 5;
  const int waveM = wave & 3;   // 4 slabs of 16 rows
  const int waveN = wave >> 2;  // 2 slabs of 64 cols
  const int mBlk  = blockIdx.y * 64;
  const int nBlk  = blockIdx.x * 128;

  const v8f vzero = {0.f, 0.f, 0.f, 0.f, 0.f, 0.f, 0.f, 0.f};
  v8f acc[4];
#pragma unroll
  for (int t = 0; t < 4; ++t) acc[t] = vzero;

  const int lr = tid >> 2;        // 0..63 tile row to load
  const int lc = (tid & 3) * 8;   // 8-wide K segment

  const int gm  = mBlk + lr;            // global X row for this loader
  const int gn0 = nBlk + lr;            // global W rows (output features)
  const int gn1 = nBlk + lr + 64;

#pragma unroll 2
  for (int k0 = 0; k0 < K; k0 += 32) {
    // ---- stage X tile (fp32 -> bf16) ----
    float4 x0, x1;
    if (gm < M) {
      const float* px = X + (size_t)gm * K + k0 + lc;
      x0 = *(const float4*)px;
      x1 = *(const float4*)(px + 4);
      if (k0 + 32 < K) __builtin_prefetch(px + 32, 0, 0);
    } else {
      x0 = make_float4(0.f, 0.f, 0.f, 0.f);
      x1 = x0;
    }
    uint4 ux;
    ux.x = pk2(x0.x, x0.y); ux.y = pk2(x0.z, x0.w);
    ux.z = pk2(x1.x, x1.y); ux.w = pk2(x1.z, x1.w);
    *(uint4*)(Xs + lr * 40 + lc) = ux;

    // ---- stage W tile, 128 rows (fp32 -> bf16) ----
    {
      const float* pw = W + (size_t)gn0 * K + k0 + lc;
      float4 w0 = *(const float4*)pw;
      float4 w1 = *(const float4*)(pw + 4);
      if (k0 + 32 < K) __builtin_prefetch(pw + 32, 0, 0);
      uint4 uw;
      uw.x = pk2(w0.x, w0.y); uw.y = pk2(w0.z, w0.w);
      uw.z = pk2(w1.x, w1.y); uw.w = pk2(w1.z, w1.w);
      *(uint4*)(Ws2 + lr * 40 + lc) = uw;
    }
    {
      const float* pw = W + (size_t)gn1 * K + k0 + lc;
      float4 w0 = *(const float4*)pw;
      float4 w1 = *(const float4*)(pw + 4);
      if (k0 + 32 < K) __builtin_prefetch(pw + 32, 0, 0);
      uint4 uw;
      uw.x = pk2(w0.x, w0.y); uw.y = pk2(w0.z, w0.w);
      uw.z = pk2(w1.x, w1.y); uw.w = pk2(w1.z, w1.w);
      *(uint4*)(Ws2 + (lr + 64) * 40 + lc) = uw;
    }

    __syncthreads();

    // ---- fragments per ISA 7.12.2 ----
    union Frag { uint4 q[2]; v16bf v; } a, b;
    // A 16x32: lanes<16 hold K{0..7,16..23}; lanes>=16 hold K{8..15,24..31}
    const int mRow  = waveM * 16 + (lane & 15);
    const int kOffA = (lane >> 4) * 8;
    a.q[0] = *(const uint4*)(Xs + mRow * 40 + kOffA);
    a.q[1] = *(const uint4*)(Xs + mRow * 40 + kOffA + 16);
    // B 32x16: lanes<16 hold K{0..7,16..23} of column n; lanes>=16 hold K{8..15,24..31}
    const int kOffB = (lane >> 4) * 8;
#pragma unroll
    for (int t = 0; t < 4; ++t) {
      const int nRow = waveN * 64 + t * 16 + (lane & 15);
      b.q[0] = *(const uint4*)(Ws2 + nRow * 40 + kOffB);
      b.q[1] = *(const uint4*)(Ws2 + nRow * 40 + kOffB + 16);
      acc[t] = __builtin_amdgcn_wmma_f32_16x16x32_bf16(
          false, a.v, false, b.v, (short)0, acc[t], false, false);
    }

    __syncthreads();
  }

  // ---- epilogue: C/D layout lanes<16 -> M=r, lanes>=16 -> M=8+r ----
  const int rowBase = mBlk + waveM * 16 + ((lane >> 4) << 3);
#pragma unroll
  for (int t = 0; t < 4; ++t) {
    const int col = nBlk + waveN * 64 + t * 16 + (lane & 15);
    const float bi = bias ? bias[col] : 0.f;
#pragma unroll
    for (int r = 0; r < 8; ++r) {
      const int row = rowBase + r;
      if (row < M) {
        float v0 = acc[t][r] + bi;
        if (relu) v0 = fmaxf(v0, 0.f);
        Y[(size_t)row * N + col] = v0;
      }
    }
  }
}

// =====================================================================
// Elementwise add
// =====================================================================
__global__ void add_kernel(const float* __restrict__ a,
                           const float* __restrict__ b,
                           float* __restrict__ y, int n)
{
  const int i = blockIdx.x * blockDim.x + threadIdx.x;
  if (i < n) y[i] = a[i] + b[i];
}

// =====================================================================
// y = LayerNorm(x + res) over last dim 256. One wave32 per row.
// =====================================================================
__global__ __launch_bounds__(256) void ln_res_kernel(
    const float* __restrict__ x, const float* __restrict__ res,
    const float* __restrict__ g, const float* __restrict__ b,
    float* __restrict__ y, int rows)
{
  const int row  = blockIdx.x * 8 + (threadIdx.x >> 5);
  const int lane = threadIdx.x & 31;
  if (row >= rows) return;
  const float* px = x   + (size_t)row * 256;
  const float* pr = res + (size_t)row * 256;
  float v[8];
  float s = 0.f, ss = 0.f;
#pragma unroll
  for (int i = 0; i < 8; ++i) {
    const int e = lane + i * 32;
    const float t = px[e] + pr[e];
    v[i] = t; s += t; ss += t * t;
  }
#pragma unroll
  for (int o = 16; o >= 1; o >>= 1) {
    s  += __shfl_xor(s,  o, 32);
    ss += __shfl_xor(ss, o, 32);
  }
  const float mean = s * (1.f / 256.f);
  const float var  = ss * (1.f / 256.f) - mean * mean;
  const float rstd = rsqrtf(var + 1e-5f);
#pragma unroll
  for (int i = 0; i < 8; ++i) {
    const int e = lane + i * 32;
    y[(size_t)row * 256 + e] = (v[i] - mean) * rstd * g[e] + b[e];
  }
}

// =====================================================================
// Self-attention core over T=25 tokens, per (b*q, head) wave.
// qk: [M_TOK, 512] (q || k), v: [M_TOK, 256], out: [M_TOK, 256]
// =====================================================================
__global__ __launch_bounds__(32) void self_attn_kernel(
    const float* __restrict__ qk, const float* __restrict__ v,
    float* __restrict__ out)
{
  __shared__ float qs[25][32], ks[25][32], vs[25][32];
  const int bq   = blockIdx.x >> 3;
  const int h    = blockIdx.x & 7;
  const int lane = threadIdx.x;

  for (int e = lane; e < 25 * 32; e += 32) {
    const int j = e >> 5, c = e & 31;
    const size_t row = (size_t)(bq * TT + j);
    qs[j][c] = qk[row * 512 + h * 32 + c];
    ks[j][c] = qk[row * 512 + 256 + h * 32 + c];
    vs[j][c] = v [row * 256 + h * 32 + c];
  }
  __syncthreads();

  if (lane < TT) {
    float p[25];
    float mx = -3.0e38f;
#pragma unroll
    for (int j = 0; j < 25; ++j) {
      float s = 0.f;
#pragma unroll
      for (int c = 0; c < 32; ++c) s += qs[lane][c] * ks[j][c];
      s *= 0.17677669529663689f;   // 1/sqrt(32)
      p[j] = s;
      mx = fmaxf(mx, s);
    }
    float sum = 0.f;
#pragma unroll
    for (int j = 0; j < 25; ++j) { p[j] = __expf(p[j] - mx); sum += p[j]; }
    const float inv = 1.f / sum;
#pragma unroll
    for (int c = 0; c < 32; ++c) {
      float o = 0.f;
#pragma unroll
      for (int j = 0; j < 25; ++j) o += p[j] * vs[j][c];
      out[(size_t)(bq * TT + lane) * 256 + h * 32 + c] = o * inv;
    }
  }
}

// =====================================================================
// MS-deformable sampling. Grid = B*LQ blocks, 8 waves = 8 heads,
// lane = channel (HD=32). Pyramid levels are square: {128,64,32,16}.
// =====================================================================
__global__ __launch_bounds__(256) void ms_deform_kernel(
    const float* __restrict__ ref, const float* __restrict__ offb,
    const float* __restrict__ awb, const float* __restrict__ val,
    float* __restrict__ out)
{
  const int row = blockIdx.x;               // b*LQ + lq
  const int b   = row / LQQ;
  const int lq  = row - b * LQQ;
  const int h   = threadIdx.x >> 5;
  const int c   = threadIdx.x & 31;

  // softmax over the 16 (level, point) weights of this head
  float p[16];
  float mx = -3.0e38f;
#pragma unroll
  for (int j = 0; j < 16; ++j) {
    p[j] = awb[(size_t)row * 128 + h * 16 + j];
    mx = fmaxf(mx, p[j]);
  }
  float ssum = 0.f;
#pragma unroll
  for (int j = 0; j < 16; ++j) { p[j] = __expf(p[j] - mx); ssum += p[j]; }
  const float inv = 1.f / ssum;

  const int dimL[4] = {128, 64, 32, 16};
  const int stL[4]  = {0, 16384, 20480, 21504};

  float acc = 0.f;
#pragma unroll
  for (int l = 0; l < 4; ++l) {
    const int HW = dimL[l];
    const float fH = (float)HW;
    const float rx = ref[((size_t)(b * LQQ + lq) * 4 + l) * 2 + 0];
    const float ry = ref[((size_t)(b * LQQ + lq) * 4 + l) * 2 + 1];
#pragma unroll
    for (int pt = 0; pt < 4; ++pt) {
      const size_t oi = (size_t)row * 256 + (size_t)(((h * 4 + l) * 4 + pt) * 2);
      const float xl = rx * fH + offb[oi]     - 0.5f;  // ref*W + off - 0.5
      const float yl = ry * fH + offb[oi + 1] - 0.5f;
      const float x0f = floorf(xl), y0f = floorf(yl);
      const float wx = xl - x0f, wy = yl - y0f;
      const int x0 = (int)x0f, y0 = (int)y0f;
      const float aww = p[l * 4 + pt] * inv;
#pragma unroll
      for (int dy = 0; dy < 2; ++dy) {
#pragma unroll
        for (int dx = 0; dx < 2; ++dx) {
          const int xi = x0 + dx, yi = y0 + dy;
          if (xi >= 0 && xi < HW && yi >= 0 && yi < HW) {
            const float cw = (dx ? wx : 1.f - wx) * (dy ? wy : 1.f - wy);
            acc += val[(size_t)(b * SS + stL[l] + yi * HW + xi) * 256 + h * 32 + c]
                   * (cw * aww);
          }
        }
      }
    }
  }
  out[(size_t)row * 256 + h * 32 + c] = acc;
}

// =====================================================================
// Orchestration
// =====================================================================
extern "C" void kernel_launch(void* const* d_in, const int* in_sizes, int n_in,
                              void* d_out, int out_size, void* d_ws, size_t ws_size,
                              hipStream_t stream)
{
  (void)in_sizes; (void)n_in; (void)out_size; (void)ws_size;

  const float* tgt        = (const float*)d_in[0];
  const float* pos        = (const float*)d_in[1];
  const float* ref        = (const float*)d_in[2];
  const float* src        = (const float*)d_in[3];
  // d_in[4], d_in[5]: spatial shapes / level starts (static, hardcoded)
  const float* in_proj_w  = (const float*)d_in[6];
  const float* in_proj_b  = (const float*)d_in[7];
  const float* out_proj_w = (const float*)d_in[8];
  const float* out_proj_b = (const float*)d_in[9];
  const float* ln1_g      = (const float*)d_in[10];
  const float* ln1_b      = (const float*)d_in[11];
  const float* so_w       = (const float*)d_in[12];
  const float* so_b       = (const float*)d_in[13];
  const float* aw_w       = (const float*)d_in[14];
  const float* aw_b       = (const float*)d_in[15];
  const float* vp_w       = (const float*)d_in[16];
  const float* vp_b       = (const float*)d_in[17];
  const float* op_w       = (const float*)d_in[18];
  const float* op_b       = (const float*)d_in[19];
  const float* ln2_g      = (const float*)d_in[20];
  const float* ln2_b      = (const float*)d_in[21];
  const float* f1_w       = (const float*)d_in[22];
  const float* f1_b       = (const float*)d_in[23];
  const float* f2_w       = (const float*)d_in[24];
  const float* f2_b       = (const float*)d_in[25];
  const float* ln3_g      = (const float*)d_in[26];
  const float* ln3_b      = (const float*)d_in[27];
  float* out = (float*)d_out;

  // ---- workspace layout (floats), regions reused over time ----
  float* ws = (float*)d_ws;
  const size_t N256 = (size_t)M_TOK * CC;   // 3,840,000
  float* R0  = ws;                  // qk_in -> tgt2 -> cross -> ffn_out
  float* R1a = ws + 1 * N256;       // qk rows 0..7499 (q||k) -> query -> tgt_ln2
  float* R1b = ws + 2 * N256;       // qk rows 7500..14999 (q||k) -> sampling offsets
  float* R2  = ws + 3 * N256;       // v -> attn-weight logits
  float* R3  = ws + 4 * N256;       // attn core out -> deform accum
  float* R4  = ws + 5 * N256;       // tgt_ln1
  float* R6  = ws + 6 * N256;       // value proj (43520*256) -> ffn hidden (15000*1024)

  const int n = (int)N256;
  const dim3 b256(256);
  const int gridM = (M_TOK + 63) / 64;       // 235
  const int gridS = (BB * SS) / 64;          // 680

  // 1. qk_in = tgt + pos
  add_kernel<<<dim3(n / 256), b256, 0, stream>>>(tgt, pos, R0, n);
  // 2. [q||k] = qk_in @ in_proj_w[0:512]^T + b  (N=512 -> contiguous R1a..R1b)
  gemm_bf16_kernel<<<dim3(4, gridM), b256, 0, stream>>>(
      R0, in_proj_w, in_proj_b, R1a, M_TOK, 512, 256, 0);
  // 3. v = tgt @ in_proj_w[512:768]^T + b
  gemm_bf16_kernel<<<dim3(2, gridM), b256, 0, stream>>>(
      tgt, in_proj_w + (size_t)512 * 256, in_proj_b + 512, R2, M_TOK, 256, 256, 0);
  // 4. attention core
  self_attn_kernel<<<dim3(BB * QQ * NHH), dim3(32), 0, stream>>>(R1a, R2, R3);
  // 5. tgt2 = attn_out @ out_proj^T + b
  gemm_bf16_kernel<<<dim3(2, gridM), b256, 0, stream>>>(
      R3, out_proj_w, out_proj_b, R0, M_TOK, 256, 256, 0);
  // 6. tgt_ln1 = LN(tgt + tgt2)
  ln_res_kernel<<<dim3(M_TOK / 8), b256, 0, stream>>>(tgt, R0, ln1_g, ln1_b, R4, M_TOK);
  // 7. query = tgt_ln1 + pos
  add_kernel<<<dim3(n / 256), b256, 0, stream>>>(R4, pos, R1a, n);
  // 8. sampling offsets (N=256)
  gemm_bf16_kernel<<<dim3(2, gridM), b256, 0, stream>>>(
      R1a, so_w, so_b, R1b, M_TOK, 256, 256, 0);
  // 9. attention-weight logits (N=128)
  gemm_bf16_kernel<<<dim3(1, gridM), b256, 0, stream>>>(
      R1a, aw_w, aw_b, R2, M_TOK, 128, 256, 0);
  // 10. value projection over the pyramid (M = B*S = 43520)
  gemm_bf16_kernel<<<dim3(2, gridS), b256, 0, stream>>>(
      src, vp_w, vp_b, R6, BB * SS, 256, 256, 0);
  // 11. deformable bilinear sampling + weight softmax -> accum
  ms_deform_kernel<<<dim3(M_TOK), b256, 0, stream>>>(ref, R1b, R2, R6, R3);
  // 12. cross = accum @ op^T + b
  gemm_bf16_kernel<<<dim3(2, gridM), b256, 0, stream>>>(
      R3, op_w, op_b, R0, M_TOK, 256, 256, 0);
  // 13. tgt_ln2 = LN(tgt_ln1 + cross)
  ln_res_kernel<<<dim3(M_TOK / 8), b256, 0, stream>>>(R4, R0, ln2_g, ln2_b, R1a, M_TOK);
  // 14. h1 = relu(tgt_ln2 @ ffn1^T + b)  (N=1024)
  gemm_bf16_kernel<<<dim3(8, gridM), b256, 0, stream>>>(
      R1a, f1_w, f1_b, R6, M_TOK, 1024, 256, 1);
  // 15. ffn_out = h1 @ ffn2^T + b  (K=1024)
  gemm_bf16_kernel<<<dim3(2, gridM), b256, 0, stream>>>(
      R6, f2_w, f2_b, R0, M_TOK, 256, 1024, 0);
  // 16. out = LN(tgt_ln2 + ffn_out)
  ln_res_kernel<<<dim3(M_TOK / 8), b256, 0, stream>>>(R1a, R0, ln3_g, ln3_b, out, M_TOK);
}
